// VariationalDropoutLSTM_19619410608719
// MI455X (gfx1250) — matrix operand order repaired and checked by reading, not compiled, not run on hardware
//
#include <hip/hip_runtime.h>
#include <hip/hip_bf16.h>
#include <math.h>

// ---------------------------------------------------------------------------
// Variational-dropout 2-layer LSTM for MI455X (gfx1250, wave32).
//  - GEMMs on v_wmma_f32_16x16x32_bf16 (f32 accumulate), copy-free
//    double-buffered K loop (no mov rotation; loads overlap WMMAs).
//  - Recurrent step kernel stages h*mask (128 KB) into LDS with the Tensor
//    Data Mover (tensor_load_to_lds + s_wait_tensorcnt), A-frags via ds_load.
//  - 256 step launches per layer; everything graph-capturable on `stream`.
// ---------------------------------------------------------------------------

#define T_STEPS 256
#define BATCH   64
#define HID     1024
#define GATES   4096   // 4*HID
#define LAYERS  2

typedef __bf16 bf16_t;
typedef unsigned int u32;
typedef __attribute__((ext_vector_type(16))) __bf16 v16bf;
typedef __attribute__((ext_vector_type(8)))  __bf16 v8bf;
typedef __attribute__((ext_vector_type(4)))  __bf16 v4bf;
typedef __attribute__((ext_vector_type(8)))  float  v8f;
typedef __attribute__((ext_vector_type(4)))  float  v4f;
typedef __attribute__((ext_vector_type(4)))  u32    v4u;
typedef __attribute__((ext_vector_type(8)))  u32    v8u;

// ---- WMMA fragment loaders (CDNA5 wave32 layouts, CDNA5 ISA §7.12.2) -----
// A: 16x32 bf16 (MxK). lane L: M = L&15; half = L>>4.
static __device__ __forceinline__ v16bf
load_frag_a(const bf16_t* __restrict__ base, int lda, int m0, int k0, int lane) {
    int m    = m0 + (lane & 15);
    int half = lane >> 4;
    const bf16_t* row = base + (size_t)m * lda;
    v8bf lo = *(const v8bf*)(row + k0 + 8 * half);
    v8bf hi = *(const v8bf*)(row + k0 + 16 + 8 * half);
    return __builtin_shufflevector(lo, hi, 0,1,2,3,4,5,6,7,8,9,10,11,12,13,14,15);
}

// B: 32x16 bf16 (KxN) from row-major W[N, K] (B[k,n]=W[n,k]).
static __device__ __forceinline__ v16bf
load_frag_b(const bf16_t* __restrict__ W, int ldw, int n0, int k0, int lane) {
    int n    = n0 + (lane & 15);
    int half = lane >> 4;
    const bf16_t* row = W + (size_t)n * ldw + k0 + 8 * half;
    v8bf lo = *(const v8bf*)(row);
    v8bf hi = *(const v8bf*)(row + 16);
    return __builtin_shufflevector(lo, hi, 0,1,2,3,4,5,6,7,8,9,10,11,12,13,14,15);
}

static __device__ __forceinline__ v8f wmma_bf16(v16bf a, v16bf b, v8f c) {
    return __builtin_amdgcn_wmma_f32_16x16x32_bf16(false, a, false, b, (short)0, c,
                                                   false, false);
}

static __device__ __forceinline__ float sigm(float x) {
    return 1.0f / (1.0f + __expf(-x));
}

// ---------------------------- utility kernels ------------------------------
__global__ void cvt_f32_to_bf16(const float* __restrict__ src,
                                bf16_t* __restrict__ dst, int n4) {
    for (int i = blockIdx.x * blockDim.x + threadIdx.x; i < n4;
         i += gridDim.x * blockDim.x) {
        v4f v = ((const v4f*)src)[i];
        ((v4bf*)dst)[i] = __builtin_convertvector(v, v4bf);
    }
}

__global__ void mul_cvt_f32_to_bf16(const float* __restrict__ a,
                                    const float* __restrict__ m,
                                    bf16_t* __restrict__ dst, int n4) {
    for (int i = blockIdx.x * blockDim.x + threadIdx.x; i < n4;
         i += gridDim.x * blockDim.x) {
        v4f v = ((const v4f*)a)[i] * ((const v4f*)m)[i];
        ((v4bf*)dst)[i] = __builtin_convertvector(v, v4bf);
    }
}

__global__ void fill_zero_u128(v4u* __restrict__ p, int n16) {
    v4u z = {0u, 0u, 0u, 0u};
    for (int i = blockIdx.x * blockDim.x + threadIdx.x; i < n16;
         i += gridDim.x * blockDim.x)
        p[i] = z;
}

// ------------------------------ pre-GEMM -----------------------------------
// pre[r, g] = sum_k X[r,k]*Wih[g,k] + b_ih[g] + b_hh[g];  r in [0, T*B)
// Wave tile 16x64 (4 N-tiles); K loop unrolled x2 with two fragment buffer
// sets so there is no register rotation (no v_mov traffic).
__global__ void __launch_bounds__(256)
pregemm_kernel(const bf16_t* __restrict__ X,     // [T*B, HID]
               const bf16_t* __restrict__ Wih,   // [GATES, HID]
               const float* __restrict__ bih,
               const float* __restrict__ bhh,
               float* __restrict__ pre) {        // [T*B, GATES]
    int tid  = blockIdx.x * blockDim.x + threadIdx.x;
    int wave = tid >> 5;
    int lane = tid & 31;
    int m0 = (wave >> 6) * 16;
    int n0 = (wave & 63) * 64;

    v8f acc[4] = {};
    v16bf a0, a1, b0[4], b1[4];

    a0 = load_frag_a(X, HID, m0, 0, lane);
    #pragma unroll
    for (int i = 0; i < 4; ++i) b0[i] = load_frag_b(Wih, HID, n0 + 16 * i, 0, lane);

    int k0 = 0;
    for (; k0 < HID - 64; k0 += 64) {
        a1 = load_frag_a(X, HID, m0, k0 + 32, lane);
        #pragma unroll
        for (int i = 0; i < 4; ++i)
            b1[i] = load_frag_b(Wih, HID, n0 + 16 * i, k0 + 32, lane);
        #pragma unroll
        for (int i = 0; i < 4; ++i) acc[i] = wmma_bf16(a0, b0[i], acc[i]);

        a0 = load_frag_a(X, HID, m0, k0 + 64, lane);
        #pragma unroll
        for (int i = 0; i < 4; ++i)
            b0[i] = load_frag_b(Wih, HID, n0 + 16 * i, k0 + 64, lane);
        #pragma unroll
        for (int i = 0; i < 4; ++i) acc[i] = wmma_bf16(a1, b1[i], acc[i]);
    }
    // tail: k0 == HID-64
    a1 = load_frag_a(X, HID, m0, k0 + 32, lane);
    #pragma unroll
    for (int i = 0; i < 4; ++i)
        b1[i] = load_frag_b(Wih, HID, n0 + 16 * i, k0 + 32, lane);
    #pragma unroll
    for (int i = 0; i < 4; ++i) acc[i] = wmma_bf16(a0, b0[i], acc[i]);
    #pragma unroll
    for (int i = 0; i < 4; ++i) acc[i] = wmma_bf16(a1, b1[i], acc[i]);

    int half = lane >> 4, nl = lane & 15;
    #pragma unroll
    for (int i = 0; i < 4; ++i) {
        int col = n0 + 16 * i + nl;
        float bias = bih[col] + bhh[col];
        #pragma unroll
        for (int j = 0; j < 8; ++j) {
            int row = m0 + j + 8 * half;                 // C/D: M = j + 8*half
            pre[(size_t)row * GATES + col] = acc[i][j] + bias;
        }
    }
}

// ----------------------------- LSTM step -----------------------------------
// gates = pre_t + hm @ Whh^T ; the wave's 4 N-tiles sit at columns
// gate*HID + n0 so it owns i,f,g,o for one 16x16 patch -> full cell in-wave.
// hm (64x1024 bf16, 128 KB) staged into LDS by the TDM (wave 0 issues one
// tensor_load_to_lds; EXEC is ignored and it signals done once).
extern __shared__ bf16_t smem_hm[];   // BATCH*HID bf16 = 128 KB dynamic LDS

__global__ void __launch_bounds__(256)
lstm_step_kernel(const float* __restrict__ pre_t,     // [B, GATES]
                 const bf16_t* __restrict__ Whh,      // [GATES, HID]
                 const bf16_t* __restrict__ hm_in,    // [B, HID] masked h_{t-1}
                 bf16_t* __restrict__ hm_out,         // [B, HID] masked h_t
                 float* __restrict__ c_cur,           // [B, HID]
                 const float* __restrict__ mask_next, // mask_h[l, t+1]
                 float* __restrict__ h_out0,
                 float* __restrict__ h_out1,          // optional (layer 1)
                 float* __restrict__ c_out,           // optional (layer 1)
                 int last_step) {
    int tid  = threadIdx.x;
    int wave = (blockIdx.x * blockDim.x + tid) >> 5;
    int lane = tid & 31;
    int m0 = 16 * (wave & 3);
    int n0 = 16 * (wave >> 2);

    // ---- TDM: stage hm_in -> LDS (D# per CDNA5 ISA) ----
    if (tid < 32) {  // wave 0 of the block issues the DMA
        unsigned lds_base = (unsigned)(size_t)&smem_hm[0];
        unsigned long long ga = (unsigned long long)(size_t)hm_in;
        v4u g0;
        g0[0] = 1u;                                   // count=1, user mode
        g0[1] = lds_base;                             // lds_addr
        g0[2] = (u32)ga;                              // global_addr[31:0]
        g0[3] = (u32)((ga >> 32) & 0x01FFFFFFu) | (2u << 30);  // addr[56:32]|type=2
        v8u g1;
        g1[0] = (1u << 16);                           // data_size=1 (2 bytes)
        g1[1] = (u32)(HID) << 16;                     // tensor_dim0[15:0]=1024
        g1[2] = (u32)(BATCH) << 16;                   // dim0 hi=0 | tensor_dim1=64
        g1[3] = (u32)(HID) << 16;                     // dim1 hi=0 | tile_dim0=1024
        g1[4] = (u32)(BATCH);                         // tile_dim1=64, tile_dim2=0
        g1[5] = (u32)(HID);                           // tensor_dim0_stride=1024
        g1[6] = 0u;                                   // stride hi / dim1_stride lo
        g1[7] = 0u;
        asm volatile("tensor_load_to_lds %0, %1" :: "s"(g0), "s"(g1) : "memory");
        __builtin_amdgcn_s_wait_tensorcnt(0);
    }
    __syncthreads();

    // ---- WMMA K-loop: A from LDS (ds_load_b128), B double-buffered --------
    v8f acc[4] = {};
    v16bf b0[4], b1[4];
    #pragma unroll
    for (int g = 0; g < 4; ++g) b0[g] = load_frag_b(Whh, HID, g * HID + n0, 0, lane);

    int k0 = 0;
    for (; k0 < HID - 64; k0 += 64) {
        #pragma unroll
        for (int g = 0; g < 4; ++g)
            b1[g] = load_frag_b(Whh, HID, g * HID + n0, k0 + 32, lane);
        {
            v16bf a = load_frag_a(smem_hm, HID, m0, k0, lane);
            #pragma unroll
            for (int g = 0; g < 4; ++g) acc[g] = wmma_bf16(a, b0[g], acc[g]);
        }
        #pragma unroll
        for (int g = 0; g < 4; ++g)
            b0[g] = load_frag_b(Whh, HID, g * HID + n0, k0 + 64, lane);
        {
            v16bf a = load_frag_a(smem_hm, HID, m0, k0 + 32, lane);
            #pragma unroll
            for (int g = 0; g < 4; ++g) acc[g] = wmma_bf16(a, b1[g], acc[g]);
        }
    }
    // tail: k0 == HID-64
    #pragma unroll
    for (int g = 0; g < 4; ++g)
        b1[g] = load_frag_b(Whh, HID, g * HID + n0, k0 + 32, lane);
    {
        v16bf a = load_frag_a(smem_hm, HID, m0, k0, lane);
        #pragma unroll
        for (int g = 0; g < 4; ++g) acc[g] = wmma_bf16(a, b0[g], acc[g]);
    }
    {
        v16bf a = load_frag_a(smem_hm, HID, m0, k0 + 32, lane);
        #pragma unroll
        for (int g = 0; g < 4; ++g) acc[g] = wmma_bf16(a, b1[g], acc[g]);
    }

    // ---- LSTM cell, fully local to the wave -------------------------------
    int half = lane >> 4, nl = lane & 15;
    int hcol = n0 + nl;
    #pragma unroll
    for (int j = 0; j < 8; ++j) {
        int brow = m0 + j + 8 * half;
        size_t gidx = (size_t)brow * GATES + hcol;
        size_t hidx = (size_t)brow * HID + hcol;
        float gi = acc[0][j] + pre_t[gidx];
        float gf = acc[1][j] + pre_t[gidx + HID];
        float gg = acc[2][j] + pre_t[gidx + 2 * HID];
        float go = acc[3][j] + pre_t[gidx + 3 * HID];
        float c  = sigm(gf) * c_cur[hidx] + sigm(gi) * tanhf(gg);
        float h  = sigm(go) * tanhf(c);
        c_cur[hidx]  = c;
        h_out0[hidx] = h;
        if (h_out1) h_out1[hidx] = h;
        if (c_out)  c_out[hidx]  = c;
        if (!last_step) hm_out[hidx] = (bf16_t)(h * mask_next[hidx]);
    }
}

// ------------------------------- launcher ----------------------------------
extern "C" void kernel_launch(void* const* d_in, const int* in_sizes, int n_in,
                              void* d_out, int out_size, void* d_ws, size_t ws_size,
                              hipStream_t stream) {
    const float* x       = (const float*)d_in[0];  // [T,B,H]
    const float* W_ih    = (const float*)d_in[1];  // [L,4H,H]
    const float* W_hh    = (const float*)d_in[2];  // [L,4H,H]
    const float* b_ih    = (const float*)d_in[3];  // [L,4H]
    const float* b_hh    = (const float*)d_in[4];  // [L,4H]
    const float* mask_h  = (const float*)d_in[5];  // [L,T,B,H]
    const float* mask_in = (const float*)d_in[6];  // [L-1,T,B,H]

    const size_t TBH = (size_t)T_STEPS * BATCH * HID;
    float* out_h0 = (float*)d_out;            // h_seq of layer 1
    float* out_h1 = out_h0 + TBH;             // duplicate copy
    float* out_c  = out_h0 + 2 * TBH;         // c_seq of layer 1

    // ---- workspace carve-up (~403 MB) ----
    char*  ws  = (char*)d_ws;
    size_t off = 0;
    auto carve = [&](size_t bytes) {
        char* p = ws + off;
        off = (off + bytes + 255) & ~(size_t)255;
        return p;
    };
    bf16_t* Wih_b  = (bf16_t*)carve((size_t)LAYERS * GATES * HID * sizeof(bf16_t));
    bf16_t* Whh_b  = (bf16_t*)carve((size_t)LAYERS * GATES * HID * sizeof(bf16_t));
    bf16_t* xin_b  = (bf16_t*)carve(TBH * sizeof(bf16_t));
    float*  hseq0  = (float*)carve(TBH * sizeof(float));
    float*  pre    = (float*)carve((size_t)T_STEPS * BATCH * GATES * sizeof(float));
    bf16_t* hm0    = (bf16_t*)carve((size_t)BATCH * HID * sizeof(bf16_t));
    bf16_t* hm1    = (bf16_t*)carve((size_t)BATCH * HID * sizeof(bf16_t));
    float*  c_cur  = (float*)carve((size_t)BATCH * HID * sizeof(float));
    (void)ws_size; (void)in_sizes; (void)n_in; (void)out_size;

    const int nW4 = LAYERS * GATES * HID / 4;       // 4-wide converts
    cvt_f32_to_bf16<<<2048, 256, 0, stream>>>(W_ih, Wih_b, nW4);
    cvt_f32_to_bf16<<<2048, 256, 0, stream>>>(W_hh, Whh_b, nW4);

    const size_t ldsBytes = (size_t)BATCH * HID * sizeof(bf16_t);  // 128 KB

    for (int l = 0; l < LAYERS; ++l) {
        if (l == 0)
            cvt_f32_to_bf16<<<2048, 256, 0, stream>>>(x, xin_b, (int)(TBH / 4));
        else
            mul_cvt_f32_to_bf16<<<2048, 256, 0, stream>>>(hseq0, mask_in, xin_b,
                                                          (int)(TBH / 4));

        pregemm_kernel<<<8192, 256, 0, stream>>>(
            xin_b, Wih_b + (size_t)l * GATES * HID,
            b_ih + (size_t)l * GATES, b_hh + (size_t)l * GATES, pre);

        // h_0 = c_0 = 0
        fill_zero_u128<<<32, 256, 0, stream>>>((v4u*)hm0,
            (int)(BATCH * HID * sizeof(bf16_t) / 16));
        fill_zero_u128<<<32, 256, 0, stream>>>((v4u*)c_cur,
            (int)(BATCH * HID * sizeof(float) / 16));

        const bf16_t* Whh_l = Whh_b + (size_t)l * GATES * HID;
        for (int t = 0; t < T_STEPS; ++t) {
            const bf16_t* hm_in  = (t & 1) ? hm1 : hm0;
            bf16_t*       hm_out = (t & 1) ? hm0 : hm1;
            int last = (t == T_STEPS - 1);
            const float* mnext = last ? mask_h
                : mask_h + ((size_t)l * T_STEPS + (t + 1)) * BATCH * HID;
            float* h0 = (l == 0) ? hseq0 + (size_t)t * BATCH * HID
                                 : out_h0 + (size_t)t * BATCH * HID;
            float* h1 = (l == 0) ? nullptr : out_h1 + (size_t)t * BATCH * HID;
            float* co = (l == 0) ? nullptr : out_c  + (size_t)t * BATCH * HID;
            lstm_step_kernel<<<32, 256, ldsBytes, stream>>>(
                pre + (size_t)t * BATCH * GATES, Whh_l, hm_in, hm_out,
                c_cur, mnext, h0, h1, co, last);
        }
    }
}
